// LSTM_74741020885467
// MI455X (gfx1250) — compile-verified, matrix operand order checked by reading, not run on hardware
//
#include <hip/hip_runtime.h>
#include <hip/hip_bf16.h>
#include <cstdint>

// LSTM forward for MI455X (gfx1250): bf16 WMMA GEMMs + fp32 state/epilogue.
//   B=64, T=256, D_IN=512, H=1024, gates G=4H=4096 in order (i,f,g,o).
// Recurrent step stages h_{t-1} into LDS with CDNA5 async-to-LDS DMA.

#define B_   64
#define T_   256
#define DIN_ 512
#define H_   1024
#define G_   4096

// h staging row stride in bf16: 1024 + 8 pad -> row base banks advance by 4,
// so a row-parallel ds_load_b128 across 16 lanes touches all 64 banks once.
#define LDSH_STRIDE 1032

typedef __attribute__((ext_vector_type(16))) __bf16 v16bf;
typedef __attribute__((ext_vector_type(8)))  __bf16 v8bf;
typedef __attribute__((ext_vector_type(8)))  float  v8f;

union V16U { v16bf v; v8bf h[2]; };

#define AS3 __attribute__((address_space(3)))
// Flat->LDS addrspace cast truncates to the 32-bit LDS byte offset.
__device__ __forceinline__ unsigned lds_byte_off(const void* p) {
  return (unsigned)(uintptr_t)(AS3 const char*)(p);
}

// --- WMMA operand loaders -------------------------------------------------
// 16-bit A-matrix 16x32 layout (ISA 7.12.2): lane L holds line (L&15); halves
// 0..7 are K = kbase + (L>=16 ? 8 : 0) + 0..7, halves 8..15 are K+16.
// B (32x16, K x N) uses the mirrored pattern where the lane's line is the
// output column n, i.e. a row of the row-major weight matrix W[n][k].
__device__ __forceinline__ v16bf load_bf16_tile(const __bf16* linebase,
                                                int kbase, unsigned lane) {
  unsigned lo = (lane & 16u) ? 8u : 0u;
  V16U u;
  u.h[0] = *(const v8bf*)(linebase + kbase + lo);        // 16B load
  u.h[1] = *(const v8bf*)(linebase + kbase + 16 + lo);   // 16B load
  return u.v;
}

__device__ __forceinline__ v16bf load_f32_as_bf16_tile(const float* __restrict__ linebase,
                                                       int kbase, unsigned lane) {
  unsigned lo = (lane & 16u) ? 8u : 0u;
  const float* p0 = linebase + kbase + lo;
  const float* p1 = p0 + 16;
  v16bf r;
#pragma unroll
  for (int i = 0; i < 8; ++i) r[i] = (__bf16)p0[i];
#pragma unroll
  for (int i = 0; i < 8; ++i) r[8 + i] = (__bf16)p1[i];
  return r;
}

__device__ __forceinline__ float sigmoid_fast(float x) {
  return 1.0f / (1.0f + __expf(-x));
}
__device__ __forceinline__ float tanh_fast(float x) {
  return 1.0f - 2.0f / (__expf(2.0f * x) + 1.0f);
}

// --- small init kernels ---------------------------------------------------
__global__ void lstm_cvt_bf16(const float* __restrict__ s, __bf16* __restrict__ d, int n) {
  int i = blockIdx.x * blockDim.x + threadIdx.x;
  if (i < n) d[i] = (__bf16)s[i];
}

__global__ void lstm_bias_sum(const float* __restrict__ a, const float* __restrict__ b,
                              float* __restrict__ o, int n) {
  int i = blockIdx.x * blockDim.x + threadIdx.x;
  if (i < n) o[i] = a[i] + b[i];
}

__global__ void lstm_zero_state(float* __restrict__ c, __bf16* __restrict__ hb) {
  int i = blockIdx.x * blockDim.x + threadIdx.x;
  if (i < B_ * H_) c[i] = 0.0f;
  if (i < 2 * B_ * H_) hb[i] = (__bf16)0.0f;
}

// --- phase 1: x_proj = x @ w_ih^T + bias  (M=B*T, N=4096, K=512) ----------
__global__ __launch_bounds__(128) void lstm_xproj_gemm(const float* __restrict__ x,
                                                       const __bf16* __restrict__ wih,
                                                       const float* __restrict__ bias,
                                                       float* __restrict__ xproj) {
  unsigned lane = threadIdx.x & 31u;
  unsigned wave = threadIdx.x >> 5;          // 0..3
  unsigned ln   = lane & 15u;
  unsigned hi8  = (lane & 16u) ? 8u : 0u;
  int m0 = (blockIdx.x * 4 + (int)wave) * 16;  // row tile base in [0, B*T)
  int n0 = blockIdx.y * 16;                    // col tile base in [0, 4096)

  const float*  arow = x   + (size_t)(m0 + ln) * DIN_;
  const __bf16* brow = wih + (size_t)(n0 + ln) * DIN_;

  v8f acc;
  float bv = bias[n0 + ln];
#pragma unroll
  for (int r = 0; r < 8; ++r) acc[r] = bv;

  for (int k = 0; k < DIN_; k += 32) {
    v16bf a = load_f32_as_bf16_tile(arow, k, lane);
    v16bf b = load_bf16_tile(brow, k, lane);
    acc = __builtin_amdgcn_wmma_f32_16x16x32_bf16(false, a, false, b, (short)0, acc,
                                                  false, false);
  }
  // C/D layout: VGPR r, lane L -> row m0+r+hi8, col n0+(L&15)
  float* obase = xproj + (size_t)m0 * G_ + n0 + ln;
#pragma unroll
  for (int r = 0; r < 8; ++r) obase[(size_t)(r + hi8) * G_] = acc[r];
}

// --- phase 2: one recurrence step -----------------------------------------
// grid.x = H/16 hidden tiles; block = 512 threads = 16 waves.
// wave = (gate g 0..3) * 4 + (batch row tile 0..3); each wave does one 16x16
// WMMA tile of gates = [x_t | h_{t-1}] @ [w_ih | w_hh]^T (+bias / +x_proj).
// h_{t-1} (64x1024 bf16, 128 KB) is staged once per block into LDS with
// global_load_async_to_lds_b128 and consumed as the WMMA A operand via DS.
template <bool XP>
__global__ __launch_bounds__(512) void lstm_step(int t,
                                                 const float*  __restrict__ xproj,
                                                 const __bf16* __restrict__ xbf,
                                                 const __bf16* __restrict__ wih,
                                                 const float*  __restrict__ bias,
                                                 const __bf16* __restrict__ whh,
                                                 const __bf16* __restrict__ hprev,
                                                 __bf16*       __restrict__ hnext,
                                                 float*        __restrict__ cstate,
                                                 float*        __restrict__ out) {
  __shared__ __bf16 ldsH[B_ * LDSH_STRIDE];  // 129 KB staged h_{t-1} (padded)
  __shared__ float  ldsG[4][B_][16];         // 16 KB gate-tile exchange

  unsigned lane = threadIdx.x & 31u;
  unsigned wave = threadIdx.x >> 5;  // 0..15
  unsigned ln   = lane & 15u;
  unsigned hi8  = (lane & 16u) ? 8u : 0u;
  int rt = (int)(wave & 3u);   // batch row tile
  int gg = (int)(wave >> 2);   // gate index (i,f,g,o)
  int m0 = rt * 16;
  int j0 = blockIdx.x * 16;    // hidden tile base
  int n  = gg * H_ + j0;       // gate-column base == weight row base

  // ---- async-stage h_{t-1} into LDS: 8192 x 16B chunks over 512 threads ----
  {
    unsigned base = lds_byte_off(&ldsH[0]);
#pragma unroll
    for (int i = 0; i < 16; ++i) {
      unsigned c   = threadIdx.x + (unsigned)i * 512u;  // chunk id
      unsigned row = c >> 7;                            // 0..63
      unsigned col = (c & 127u) * 8u;                   // bf16 element in row
      uint64_t ga  = (uint64_t)(hprev + (size_t)row * H_ + col);
      unsigned la  = base + (row * LDSH_STRIDE + col) * 2u;
      asm volatile("global_load_async_to_lds_b128 %0, %1, off"
                   :: "v"(la), "v"(ga) : "memory");
    }
    asm volatile("s_wait_asynccnt 0x0" ::: "memory");
  }
  __syncthreads();

  v8f acc;
  if (XP) {
#pragma unroll
    for (int r = 0; r < 8; ++r) {
      int b = m0 + r + (int)hi8;
      acc[r] = xproj[((size_t)b * T_ + t) * G_ + n + ln];
    }
  } else {
    float bv = bias[n + ln];
#pragma unroll
    for (int r = 0; r < 8; ++r) acc[r] = bv;
    const __bf16* arow = xbf + ((size_t)(m0 + ln) * T_ + t) * DIN_;
    const __bf16* brow = wih + (size_t)(n + ln) * DIN_;
    for (int k = 0; k < DIN_; k += 32) {
      v16bf a = load_bf16_tile(arow, k, lane);
      v16bf b = load_bf16_tile(brow, k, lane);
      acc = __builtin_amdgcn_wmma_f32_16x16x32_bf16(false, a, false, b, (short)0, acc,
                                                    false, false);
    }
  }

  {  // recurrent part: h_{t-1} @ w_hh^T, K = 1024; A from LDS, B from L2
    const __bf16* arow = &ldsH[(unsigned)(m0 + ln) * LDSH_STRIDE];
    const __bf16* brow = whh + (size_t)(n + ln) * H_;
    for (int k = 0; k < H_; k += 32) {
      v16bf a = load_bf16_tile(arow, k, lane);   // ds_load_b128 x2
      v16bf b = load_bf16_tile(brow, k, lane);   // global_load_b128 x2
      acc = __builtin_amdgcn_wmma_f32_16x16x32_bf16(false, a, false, b, (short)0, acc,
                                                    false, false);
    }
  }

#pragma unroll
  for (int r = 0; r < 8; ++r) ldsG[gg][m0 + r + (int)hi8][ln] = acc[r];
  __syncthreads();

  // Epilogue: 64 rows x 16 hidden cols = 1024 elements over 512 threads.
  for (int e = (int)threadIdx.x; e < B_ * 16; e += 512) {
    int b  = e >> 4;
    int cc = e & 15;
    int j  = j0 + cc;
    float ig = sigmoid_fast(ldsG[0][b][cc]);
    float fg = sigmoid_fast(ldsG[1][b][cc]);
    float gt = tanh_fast   (ldsG[2][b][cc]);
    float og = sigmoid_fast(ldsG[3][b][cc]);
    size_t ci = (size_t)b * H_ + j;
    float cn = fg * cstate[ci] + ig * gt;
    cstate[ci] = cn;
    float hv = og * tanh_fast(cn);
    out[((size_t)b * T_ + t) * H_ + j] = hv;
    hnext[ci] = (__bf16)hv;
  }
}

// --- host launch ----------------------------------------------------------
extern "C" void kernel_launch(void* const* d_in, const int* in_sizes, int n_in,
                              void* d_out, int out_size, void* d_ws, size_t ws_size,
                              hipStream_t stream) {
  (void)in_sizes; (void)n_in; (void)out_size;
  const float* x    = (const float*)d_in[0];
  const float* w_ih = (const float*)d_in[1];
  const float* w_hh = (const float*)d_in[2];
  const float* b_ih = (const float*)d_in[3];
  const float* b_hh = (const float*)d_in[4];
  float* out = (float*)d_out;

  char* p = (char*)d_ws;
  auto take = [&](size_t bytes) {
    char* r = p;
    p += (bytes + 255) & ~(size_t)255;
    return r;
  };
  __bf16* wih_bf = (__bf16*)take((size_t)G_ * DIN_ * 2);
  __bf16* whh_bf = (__bf16*)take((size_t)G_ * H_ * 2);
  float*  bias   = (float*) take((size_t)G_ * 4);
  float*  cbuf   = (float*) take((size_t)B_ * H_ * 4);
  __bf16* hbuf   = (__bf16*)take((size_t)2 * B_ * H_ * 2);
  size_t base_used = (size_t)(p - (char*)d_ws);
  size_t xproj_bytes = (size_t)B_ * T_ * G_ * 4;  // 268 MB
  bool use_xproj = ws_size >= base_used + xproj_bytes;
  float*  xproj = (float*)p;
  __bf16* xbf   = (__bf16*)p;  // fallback path reuses the tail of ws

  lstm_cvt_bf16<<<(G_ * DIN_ + 255) / 256, 256, 0, stream>>>(w_ih, wih_bf, G_ * DIN_);
  lstm_cvt_bf16<<<(G_ * H_ + 255) / 256, 256, 0, stream>>>(w_hh, whh_bf, G_ * H_);
  lstm_bias_sum<<<(G_ + 255) / 256, 256, 0, stream>>>(b_ih, b_hh, bias, G_);
  lstm_zero_state<<<(2 * B_ * H_ + 255) / 256, 256, 0, stream>>>(cbuf, hbuf);

  if (use_xproj) {
    dim3 g(B_ * T_ / 64, G_ / 16);  // (256, 256)
    lstm_xproj_gemm<<<g, 128, 0, stream>>>(x, wih_bf, bias, xproj);
  } else {
    lstm_cvt_bf16<<<(B_ * T_ * DIN_ + 255) / 256, 256, 0, stream>>>(x, xbf,
                                                                    B_ * T_ * DIN_);
  }

  for (int t = 0; t < T_; ++t) {
    __bf16* hp = hbuf + (size_t)(t & 1) * B_ * H_;
    __bf16* hn = hbuf + (size_t)((t + 1) & 1) * B_ * H_;
    if (use_xproj) {
      lstm_step<true><<<H_ / 16, 512, 0, stream>>>(t, xproj, nullptr, nullptr, nullptr,
                                                   whh_bf, hp, hn, cbuf, out);
    } else {
      lstm_step<false><<<H_ / 16, 512, 0, stream>>>(t, nullptr, xbf, wih_bf, bias,
                                                    whh_bf, hp, hn, cbuf, out);
    }
  }
}
